// QuantizedLinear_14551349199601
// MI455X (gfx1250) — matrix operand order repaired and checked by reading, not compiled, not run on hardware
//
#include <hip/hip_runtime.h>

typedef __attribute__((ext_vector_type(16))) _Float16 v16h;
typedef __attribute__((ext_vector_type(8)))  _Float16 v8h;
typedef __attribute__((ext_vector_type(4)))  _Float16 v4h;
typedef __attribute__((ext_vector_type(8)))  float    v8f;
typedef __attribute__((ext_vector_type(4)))  float    f32x4;
typedef __attribute__((ext_vector_type(4)))  int      i32x4;

#define AS1 __attribute__((address_space(1)))
#define AS3 __attribute__((address_space(3)))

namespace {
constexpr int MTOT = 4 * 2048;   // B*S
constexpr int NTOT = 11008;      // OUT
constexpr int KTOT = 4096;       // IN
// phase-2 GEMM tiling
constexpr int BM = 128, BN = 128, BK = 64;
constexpr int LDT = BK + 8;      // 72 halves = 144 B padded row stride
constexpr int NKT = KTOT / BK;   // 64
// fused-fallback tiling
constexpr int FBK = 32, FLDT = FBK + 8, FNKT = KTOT / FBK;
constexpr size_t WS_NEED = ((size_t)NTOT * KTOT + (size_t)MTOT * KTOT) * 2;
}

#if defined(__gfx1250__) && __has_builtin(__builtin_amdgcn_global_load_async_to_lds_b128)
#define HAVE_ASYNC_LDS 1
#else
#define HAVE_ASYNC_LDS 0
#endif

__device__ __forceinline__ void wait_async_all() {
#if defined(__gfx1250__) && __has_builtin(__builtin_amdgcn_s_wait_asynccnt)
  __builtin_amdgcn_s_wait_asynccnt(0);
#elif HAVE_ASYNC_LDS
  asm volatile("s_wait_asynccnt 0x0" ::: "memory");
#endif
}

__device__ __forceinline__ void async_cp16(const _Float16* g, _Float16* l) {
#if HAVE_ASYNC_LDS
  // signature (from clang diagnostic): (i32x4 AS1*, i32x4 AS3*, i32 offset, i32 cpol)
  __builtin_amdgcn_global_load_async_to_lds_b128(
      (AS1 i32x4*)g, (AS3 i32x4*)l, 0, 0);
#else
  *(v8h*)l = *(const v8h*)g;   // fallback: vmem load + ds_store
#endif
}

__device__ __forceinline__ v16h cat16(v8h lo, v8h hi) {
  v16h r;
#pragma unroll
  for (int i = 0; i < 8; ++i) { r[i] = lo[i]; r[i + 8] = hi[i]; }
  return r;
}

// ---------------- phase 1: one-time dtype conversion into d_ws ----------------

__global__ __launch_bounds__(256) void cvt_w_kernel(const int* __restrict__ w,
                                                    _Float16* __restrict__ dst) {
  const size_t i = ((size_t)blockIdx.x * 256 + threadIdx.x) * 8;
  const i32x4 a = __builtin_nontemporal_load((const i32x4*)(w + i));
  const i32x4 b = __builtin_nontemporal_load((const i32x4*)(w + i + 4));
  v8h h;
#pragma unroll
  for (int j = 0; j < 4; ++j) {
    h[j]     = (_Float16)(float)a[j];   // int8 range: exact in fp16
    h[j + 4] = (_Float16)(float)b[j];
  }
  *(v8h*)(dst + i) = h;
}

__global__ __launch_bounds__(256) void cvt_x_kernel(const float* __restrict__ x,
                                                    _Float16* __restrict__ dst) {
  const size_t i = ((size_t)blockIdx.x * 256 + threadIdx.x) * 8;
  const f32x4 a = __builtin_nontemporal_load((const f32x4*)(x + i));
  const f32x4 b = __builtin_nontemporal_load((const f32x4*)(x + i + 4));
  v8h h;
#pragma unroll
  for (int j = 0; j < 4; ++j) {
    h[j]     = (_Float16)a[j];
    h[j + 4] = (_Float16)b[j];
  }
  *(v8h*)(dst + i) = h;
}

// ---------------- phase 2: fp16 WMMA GEMM (operands L2-resident) ----------------

__global__ __launch_bounds__(256, 2)
void gemm_f16_wmma(const _Float16* __restrict__ Ax,   // x  fp16 [M][K]
                   const _Float16* __restrict__ Bw,   // W  fp16 [N][K]
                   const float* __restrict__ scale,
                   const float* __restrict__ bias,
                   float* __restrict__ out) {
  __shared__ _Float16 Als[2][BM * LDT];
  __shared__ _Float16 Bls[2][BN * LDT];

  const int tid  = threadIdx.x;
  const int lane = tid & 31;
  const int wave = tid >> 5;
  const int wm   = wave & 1;
  const int wn   = wave >> 1;
  const int hh   = lane >> 4;
  const int lr   = lane & 15;

  const int bm0 = blockIdx.y * BM;
  const int bn0 = blockIdx.x * BN;

  // staging map: thread t copies 64B of one A row and one B row (4x b128 each)
  const int cr = tid >> 1;          // row 0..127
  const int cs = (tid & 1) * 32;    // half-offset within row (0 or 32)

  auto stage = [&](int buf, int k0) {
    const _Float16* gA = Ax + (size_t)(bm0 + cr) * KTOT + k0 + cs;
    const _Float16* gB = Bw + (size_t)(bn0 + cr) * KTOT + k0 + cs;
    _Float16* lA = &Als[buf][cr * LDT + cs];
    _Float16* lB = &Bls[buf][cr * LDT + cs];
#pragma unroll
    for (int j = 0; j < 4; ++j) {
      async_cp16(gA + j * 8, lA + j * 8);
      async_cp16(gB + j * 8, lB + j * 8);
    }
  };

  const v8f vzero = {};
  v8f acc[4][2];
#pragma unroll
  for (int a = 0; a < 4; ++a)
#pragma unroll
    for (int b = 0; b < 2; ++b) acc[a][b] = vzero;

  stage(0, 0);

#pragma unroll 1
  for (int kt = 0; kt < NKT; ++kt) {
    wait_async_all();              // own async copies for buf (kt&1) complete
    __syncthreads();               // everyone's complete; prev readers done
    if (kt + 1 < NKT) stage((kt + 1) & 1, (kt + 1) * BK);

    const _Float16* A = Als[kt & 1];
    const _Float16* B = Bls[kt & 1];
#pragma unroll
    for (int ks = 0; ks < 2; ++ks) {           // two 16x16x32 k-steps per tile
      v16h bf[2];
#pragma unroll
      for (int fn = 0; fn < 2; ++fn) {
        const _Float16* p = &B[(wn * 32 + fn * 16 + lr) * LDT + ks * 32 + hh * 16];
        bf[fn] = cat16(*(const v8h*)p, *(const v8h*)(p + 8));
      }
#pragma unroll
      for (int fm = 0; fm < 4; ++fm) {
        const _Float16* p = &A[(wm * 64 + fm * 16 + lr) * LDT + ks * 32 + hh * 16];
        const v16h af = cat16(*(const v8h*)p, *(const v8h*)(p + 8));
#pragma unroll
        for (int fn = 0; fn < 2; ++fn) {
          acc[fm][fn] = __builtin_amdgcn_wmma_f32_16x16x32_f16(
              false, af, false, bf[fn], (short)0, acc[fm][fn], false, false);
        }
      }
    }
  }

  // epilogue: y = scale[n]*acc + bias[n]; NT stores keep operands in L2
#pragma unroll
  for (int fn = 0; fn < 2; ++fn) {
    const int n = bn0 + wn * 32 + fn * 16 + lr;
    const float sc = scale[n];
    const float bi = bias[n];
#pragma unroll
    for (int fm = 0; fm < 4; ++fm) {
      const int mbase = bm0 + wm * 64 + fm * 16 + hh * 8;
#pragma unroll
      for (int r = 0; r < 8; ++r) {
        __builtin_nontemporal_store(acc[fm][fn][r] * sc + bi,
                                    &out[(size_t)(mbase + r) * NTOT + n]);
      }
    }
  }
}

// ---------------- fused fallback (if d_ws too small) ----------------

__global__ __launch_bounds__(256, 2)
void qlinear_fused(const float* __restrict__ x, const int* __restrict__ w,
                   const float* __restrict__ scale, const float* __restrict__ bias,
                   float* __restrict__ out) {
  __shared__ _Float16 Als[2][BM * FLDT];
  __shared__ _Float16 Bls[2][BN * FLDT];

  const int tid = threadIdx.x, lane = tid & 31, wave = tid >> 5;
  const int wm = wave & 1, wn = wave >> 1, hh = lane >> 4, lr = lane & 15;
  const int bm0 = blockIdx.y * BM, bn0 = blockIdx.x * BN;
  const int c4 = (tid & 7) * 4, r0 = tid >> 3;

  f32x4 xr[4];
  i32x4 wr[4];
  auto gload = [&](int k0) {
#pragma unroll
    for (int i = 0; i < 4; ++i) {
      const int r = r0 + 32 * i;
      xr[i] = *(const f32x4*)(x + (size_t)(bm0 + r) * KTOT + k0 + c4);
      wr[i] = *(const i32x4*)(w + (size_t)(bn0 + r) * KTOT + k0 + c4);
    }
  };
  auto stage = [&](int buf) {
#pragma unroll
    for (int i = 0; i < 4; ++i) {
      const int r = r0 + 32 * i;
      v4h ha, hb;
#pragma unroll
      for (int j = 0; j < 4; ++j) {
        ha[j] = (_Float16)xr[i][j];
        hb[j] = (_Float16)(float)wr[i][j];
      }
      *(v4h*)(&Als[buf][r * FLDT + c4]) = ha;
      *(v4h*)(&Bls[buf][r * FLDT + c4]) = hb;
    }
  };

  const v8f vzero = {};
  v8f acc[4][2];
#pragma unroll
  for (int a = 0; a < 4; ++a)
#pragma unroll
    for (int b = 0; b < 2; ++b) acc[a][b] = vzero;

  gload(0);
  stage(0);
#pragma unroll 1
  for (int kt = 0; kt < FNKT; ++kt) {
    __syncthreads();
    if (kt + 1 < FNKT) gload((kt + 1) * FBK);
    const _Float16* A = Als[kt & 1];
    const _Float16* B = Bls[kt & 1];
    v16h bf[2];
#pragma unroll
    for (int fn = 0; fn < 2; ++fn) {
      const _Float16* p = &B[(wn * 32 + fn * 16 + lr) * FLDT + hh * 16];
      bf[fn] = cat16(*(const v8h*)p, *(const v8h*)(p + 8));
    }
#pragma unroll
    for (int fm = 0; fm < 4; ++fm) {
      const _Float16* p = &A[(wm * 64 + fm * 16 + lr) * FLDT + hh * 16];
      const v16h af = cat16(*(const v8h*)p, *(const v8h*)(p + 8));
#pragma unroll
      for (int fn = 0; fn < 2; ++fn) {
        acc[fm][fn] = __builtin_amdgcn_wmma_f32_16x16x32_f16(
            false, af, false, bf[fn], (short)0, acc[fm][fn], false, false);
      }
    }
    if (kt + 1 < FNKT) stage((kt + 1) & 1);
  }
#pragma unroll
  for (int fn = 0; fn < 2; ++fn) {
    const int n = bn0 + wn * 32 + fn * 16 + lr;
    const float sc = scale[n];
    const float bi = bias[n];
#pragma unroll
    for (int fm = 0; fm < 4; ++fm) {
      const int mbase = bm0 + wm * 64 + fm * 16 + hh * 8;
#pragma unroll
      for (int r = 0; r < 8; ++r) {
        __builtin_nontemporal_store(acc[fm][fn][r] * sc + bi,
                                    &out[(size_t)(mbase + r) * NTOT + n]);
      }
    }
  }
}

extern "C" void kernel_launch(void* const* d_in, const int* in_sizes, int n_in,
                              void* d_out, int out_size, void* d_ws, size_t ws_size,
                              hipStream_t stream) {
  (void)in_sizes; (void)n_in; (void)out_size;
  const float* x     = (const float*)d_in[0];
  const int*   w8    = (const int*)  d_in[1];
  const float* scale = (const float*)d_in[2];
  const float* bias  = (const float*)d_in[3];
  float*       out   = (float*)d_out;

  if (ws_size >= WS_NEED) {
    _Float16* wsW = (_Float16*)d_ws;                       // [N][K] fp16
    _Float16* wsX = wsW + (size_t)NTOT * KTOT;             // [M][K] fp16
    cvt_w_kernel<<<(int)((size_t)NTOT * KTOT / 2048), 256, 0, stream>>>(w8, wsW);
    cvt_x_kernel<<<(int)((size_t)MTOT * KTOT / 2048), 256, 0, stream>>>(x, wsX);
    dim3 grid(NTOT / BN, MTOT / BM);   // x fastest over N: W stays L2-resident
    gemm_f16_wmma<<<grid, 256, 0, stream>>>(wsX, wsW, scale, bias, out);
  } else {
    dim3 grid(NTOT / BN, MTOT / BM);
    qlinear_fused<<<grid, 256, 0, stream>>>(x, w8, scale, bias, out);
  }
}
